// DecoderRNN_38259568673336
// MI455X (gfx1250) — matrix operand order repaired and checked by reading, not compiled, not run on hardware
//
#include <hip/hip_runtime.h>

#define H   1024
#define H3  3072
#define V   32000
#define B   32
#define T   64

typedef __attribute__((ext_vector_type(16))) __bf16 v16bf;
typedef __attribute__((ext_vector_type(8)))  float  v8f;

union FragB16 { uint4 u[2]; v16bf v; };

__device__ __forceinline__ unsigned short f32_to_bf16(float f) {
  unsigned int x = __float_as_uint(f);
  unsigned int lsb = (x >> 16) & 1u;
  x += 0x7fffu + lsb;                       // round-to-nearest-even
  return (unsigned short)(x >> 16);
}

// ---------------- one-time setup kernels ----------------

__global__ void drnn_cvt_bf16(const float* __restrict__ src,
                              unsigned short* __restrict__ dst, int n) {
  for (int i = blockIdx.x * blockDim.x + threadIdx.x; i < n;
       i += gridDim.x * blockDim.x)
    dst[i] = f32_to_bf16(src[i]);
}

__global__ void drnn_init_state(const float* __restrict__ h0,
                                float* __restrict__ h_f,
                                unsigned short* __restrict__ h_bf,
                                int* __restrict__ tok) {
  int i = blockIdx.x * blockDim.x + threadIdx.x;
  if (i < B * H) { float v = h0[i]; h_f[i] = v; h_bf[i] = f32_to_bf16(v); }
  if (i < B) tok[i] = 0;   // SOS
}

// ---------------- per-step kernels ----------------

__global__ void drnn_embed_relu(const float* __restrict__ emb,
                                const int* __restrict__ tok,
                                unsigned short* __restrict__ x_bf) {
  int i = blockIdx.x * blockDim.x + threadIdx.x;
  if (i >= B * H) return;
  int b = i >> 10, j = i & (H - 1);
  float f = emb[(size_t)tok[b] * H + j];
  x_bf[i] = f32_to_bf16(fmaxf(f, 0.0f));
}

// gi = x @ W_ih^T + b_ih ; gh = h @ W_hh^T + b_hh     [32 x 3072] each
// grid = 48 blocks, 256 threads (8 waves, 2Mx4N tile arrangement)
__global__ void __launch_bounds__(256)
drnn_gru_gemm(const unsigned short* __restrict__ x_bf,
              const unsigned short* __restrict__ h_bf,
              const unsigned short* __restrict__ Wih,
              const unsigned short* __restrict__ Whh,
              const float* __restrict__ b_ih, const float* __restrict__ b_hh,
              float* __restrict__ gi, float* __restrict__ gh) {
  const int tid  = threadIdx.x;
  const int wave = tid >> 5, lane = tid & 31, half = lane >> 4, l = lane & 15;
  const int mBase = (wave & 1) * 16;
  const int nBase = blockIdx.x * 64 + (wave >> 1) * 16;
  const unsigned short* xp  = x_bf + (size_t)(mBase + l) * H;
  const unsigned short* hp  = h_bf + (size_t)(mBase + l) * H;
  const unsigned short* wip = Wih  + (size_t)(nBase + l) * H;
  const unsigned short* whp = Whh  + (size_t)(nBase + l) * H;
  v8f ci = {}, ch = {};
  for (int kk = 0; kk < H; kk += 32) {
    FragB16 ax, ah, bi, bh;
    const int aOff = kk + half * 16;         // A layout: 16 contiguous K per half
    ax.u[0] = *(const uint4*)(xp + aOff);
    ax.u[1] = *(const uint4*)(xp + aOff + 8);
    ah.u[0] = *(const uint4*)(hp + aOff);
    ah.u[1] = *(const uint4*)(hp + aOff + 8);
    const int bOff = kk + half * 16;         // B layout: 16 contiguous K per half
    bi.u[0] = *(const uint4*)(wip + bOff);
    bi.u[1] = *(const uint4*)(wip + bOff + 8);
    bh.u[0] = *(const uint4*)(whp + bOff);
    bh.u[1] = *(const uint4*)(whp + bOff + 8);
    ci = __builtin_amdgcn_wmma_f32_16x16x32_bf16(false, ax.v, false, bi.v,
                                                 (short)0, ci, false, false);
    ch = __builtin_amdgcn_wmma_f32_16x16x32_bf16(false, ah.v, false, bh.v,
                                                 (short)0, ch, false, false);
  }
  const int n  = nBase + l;
  const float bin = b_ih[n], bhn = b_hh[n];
  const int m0 = mBase + (half ? 8 : 0);
#pragma unroll
  for (int i = 0; i < 8; ++i) {
    gi[(size_t)(m0 + i) * H3 + n] = ci[i] + bin;
    gh[(size_t)(m0 + i) * H3 + n] = ch[i] + bhn;
  }
}

__global__ void drnn_gates(const float* __restrict__ gi,
                           const float* __restrict__ gh,
                           float* __restrict__ h_f,
                           unsigned short* __restrict__ h_bf) {
  int i = blockIdx.x * blockDim.x + threadIdx.x;
  if (i >= B * H) return;
  int b = i >> 10, j = i & (H - 1);
  const float* gib = gi + (size_t)b * H3;
  const float* ghb = gh + (size_t)b * H3;
  float r = 1.0f / (1.0f + expf(-(gib[j] + ghb[j])));
  float z = 1.0f / (1.0f + expf(-(gib[H + j] + ghb[H + j])));
  float n = tanhf(gib[2 * H + j] + r * ghb[2 * H + j]);
  float hn = (1.0f - z) * n + z * h_f[i];
  h_f[i]  = hn;
  h_bf[i] = f32_to_bf16(hn);
}

// logits = h @ W_out^T + b_out, written directly into d_out[b][t][v]
// grid = 250 blocks x 256 threads; block tile = 32(M) x 128(N);
// each wave computes two adjacent 16x16 N-tiles sharing one A fragment.
// h (64 KB bf16) staged into LDS via the CDNA5 async direct-to-LDS path.
__global__ void __launch_bounds__(256)
drnn_logits_gemm(const unsigned short* __restrict__ h_bf,
                 const unsigned short* __restrict__ Wout,
                 const float* __restrict__ b_out,
                 float* __restrict__ out /* = d_out + t*V */) {
  __shared__ uint4 hl4[B * H / 8];          // 64 KB
  const int tid = threadIdx.x;
  const uint4* hsrc = (const uint4*)h_bf;
  for (int i = tid; i < B * H / 8; i += 256) {
    unsigned ldsa = (unsigned)(size_t)(&hl4[i]);   // flat->LDS: addr[31:0]
    const uint4* g = hsrc + i;
    asm volatile("global_load_async_to_lds_b128 %0, %1, off"
                 :: "v"(ldsa), "v"(g) : "memory");
  }
  asm volatile("s_wait_asynccnt 0x0" ::: "memory");
  __syncthreads();
  const unsigned short* hl = (const unsigned short*)hl4;

  const int wave = tid >> 5, lane = tid & 31, half = lane >> 4, l = lane & 15;
  const int mBase  = (wave & 1) * 16;
  const int nBase0 = blockIdx.x * 128 + (wave >> 1) * 32;   // two tiles: +0, +16
  const unsigned short* ap  = hl   + (size_t)(mBase + l) * H;
  const unsigned short* bp0 = Wout + (size_t)(nBase0 + l) * H;
  const unsigned short* bp1 = Wout + (size_t)(nBase0 + 16 + l) * H;
  v8f c0 = {}, c1 = {};
  for (int kk = 0; kk < H; kk += 32) {
    FragB16 a, b0, b1;
    const int aOff = kk + half * 16;
    a.u[0] = *(const uint4*)(ap + aOff);
    a.u[1] = *(const uint4*)(ap + aOff + 8);
    const int bOff = kk + half * 16;
    b0.u[0] = *(const uint4*)(bp0 + bOff);
    b0.u[1] = *(const uint4*)(bp0 + bOff + 8);
    b1.u[0] = *(const uint4*)(bp1 + bOff);
    b1.u[1] = *(const uint4*)(bp1 + bOff + 8);
    c0 = __builtin_amdgcn_wmma_f32_16x16x32_bf16(false, a.v, false, b0.v,
                                                 (short)0, c0, false, false);
    c1 = __builtin_amdgcn_wmma_f32_16x16x32_bf16(false, a.v, false, b1.v,
                                                 (short)0, c1, false, false);
  }
  const int n0 = nBase0 + l, n1 = nBase0 + 16 + l;
  const float bias0 = b_out[n0], bias1 = b_out[n1];
  const int m0 = mBase + (half ? 8 : 0);
#pragma unroll
  for (int i = 0; i < 8; ++i) {
    out[(size_t)(m0 + i) * ((size_t)T * V) + n0] = c0[i] + bias0;
    out[(size_t)(m0 + i) * ((size_t)T * V) + n1] = c1[i] + bias1;
  }
}

// greedy argmax over one step's logits; first-index tie-break (matches jnp.argmax)
__global__ void drnn_argmax(const float* __restrict__ out /* d_out + t*V */,
                            int* __restrict__ tok) {
  __shared__ float sv[256];
  __shared__ int   si[256];
  const int b = blockIdx.x, tid = threadIdx.x;
  const float* row = out + (size_t)b * T * V;
  float best = -__builtin_inff(); int bidx = 0;
  for (int v = tid; v < V; v += 256) {
    float f = row[v];
    if (f > best) { best = f; bidx = v; }
  }
  sv[tid] = best; si[tid] = bidx;
  __syncthreads();
  for (int s = 128; s > 0; s >>= 1) {
    if (tid < s) {
      float f2 = sv[tid + s]; int i2 = si[tid + s];
      if (f2 > sv[tid] || (f2 == sv[tid] && i2 < si[tid])) {
        sv[tid] = f2; si[tid] = i2;
      }
    }
    __syncthreads();
  }
  if (tid == 0) tok[b] = si[0];
}

// in-place log_softmax over each [V] row of d_out (B*T rows)
__global__ void drnn_logsoftmax(float* __restrict__ out) {
  __shared__ float red[256];
  const int tid = threadIdx.x;
  float* row = out + (size_t)blockIdx.x * V;
  float m = -__builtin_inff();
  for (int v = tid; v < V; v += 256) m = fmaxf(m, row[v]);
  red[tid] = m; __syncthreads();
  for (int s = 128; s > 0; s >>= 1) {
    if (tid < s) red[tid] = fmaxf(red[tid], red[tid + s]);
    __syncthreads();
  }
  m = red[0]; __syncthreads();
  float sum = 0.0f;
  for (int v = tid; v < V; v += 256) sum += expf(row[v] - m);
  red[tid] = sum; __syncthreads();
  for (int s = 128; s > 0; s >>= 1) {
    if (tid < s) red[tid] += red[tid + s];
    __syncthreads();
  }
  float lse = m + logf(red[0]);
  for (int v = tid; v < V; v += 256) row[v] = row[v] - lse;
}

__global__ void drnn_copy_hT(const float* __restrict__ h_f,
                             float* __restrict__ dst) {
  int i = blockIdx.x * blockDim.x + threadIdx.x;
  if (i < B * H) dst[i] = h_f[i];
}

// ---------------- host launcher ----------------

extern "C" void kernel_launch(void* const* d_in, const int* in_sizes, int n_in,
                              void* d_out, int out_size, void* d_ws, size_t ws_size,
                              hipStream_t stream) {
  (void)in_sizes; (void)n_in; (void)out_size; (void)ws_size;
  const float* enc_hidden = (const float*)d_in[1];   // [1,B,H]
  const float* emb        = (const float*)d_in[2];   // [V,H]
  const float* W_ih       = (const float*)d_in[3];   // [3H,H]
  const float* W_hh       = (const float*)d_in[4];   // [3H,H]
  const float* b_ih       = (const float*)d_in[5];   // [3H]
  const float* b_hh       = (const float*)d_in[6];   // [3H]
  const float* W_out      = (const float*)d_in[7];   // [V,H]
  const float* b_out      = (const float*)d_in[8];   // [V]
  float* out = (float*)d_out;                        // [B,T,V] ++ [1,B,H]

  char* ws = (char*)d_ws;
  size_t off = 0;
  unsigned short* Wih_bf  = (unsigned short*)(ws + off); off += (size_t)H3 * H * 2;
  unsigned short* Whh_bf  = (unsigned short*)(ws + off); off += (size_t)H3 * H * 2;
  unsigned short* Wout_bf = (unsigned short*)(ws + off); off += (size_t)V * H * 2;
  unsigned short* x_bf    = (unsigned short*)(ws + off); off += (size_t)B * H * 2;
  unsigned short* h_bf    = (unsigned short*)(ws + off); off += (size_t)B * H * 2;
  float* h_f = (float*)(ws + off); off += (size_t)B * H * 4;
  float* gi  = (float*)(ws + off); off += (size_t)B * H3 * 4;
  float* gh  = (float*)(ws + off); off += (size_t)B * H3 * 4;
  int*   tok = (int*)(ws + off);   off += (size_t)B * 4;

  // one-time (per call) weight conversion: bf16 weights keep W_out resident in L2
  drnn_cvt_bf16<<<1024, 256, 0, stream>>>(W_ih, Wih_bf, H3 * H);
  drnn_cvt_bf16<<<1024, 256, 0, stream>>>(W_hh, Whh_bf, H3 * H);
  drnn_cvt_bf16<<<4096, 256, 0, stream>>>(W_out, Wout_bf, V * H);
  drnn_init_state<<<(B * H + 255) / 256, 256, 0, stream>>>(enc_hidden, h_f, h_bf, tok);

  for (int t = 0; t < T; ++t) {
    float* out_t = out + (size_t)t * V;
    drnn_embed_relu<<<(B * H + 255) / 256, 256, 0, stream>>>(emb, tok, x_bf);
    drnn_gru_gemm<<<H3 / 64, 256, 0, stream>>>(x_bf, h_bf, Wih_bf, Whh_bf,
                                               b_ih, b_hh, gi, gh);
    drnn_gates<<<(B * H + 255) / 256, 256, 0, stream>>>(gi, gh, h_f, h_bf);
    drnn_logits_gemm<<<V / 128, 256, 0, stream>>>(h_bf, Wout_bf, b_out, out_t);
    drnn_argmax<<<B, 256, 0, stream>>>(out_t, tok);
  }

  drnn_logsoftmax<<<B * T, 256, 0, stream>>>(out);
  drnn_copy_hT<<<(B * H + 255) / 256, 256, 0, stream>>>(h_f, out + (size_t)B * T * V);
}
